// MambaVisualEncoder_49615462203571
// MI455X (gfx1250) — matrix operand order repaired and checked by reading, not compiled, not run on hardware
//
#include <hip/hip_runtime.h>
#include <hip/hip_bf16.h>
#include <cstdint>
#include <cstddef>

// Problem constants (from reference)
#define BB 2
#define IMG 512
#define PS 16
#define DD 384
#define EE 768
#define SS 16
#define KK 4
#define LL 12
#define GG 32
#define NN 1024
#define MM (BB*NN)   // 2048 rows
#define KMAX 768     // largest GEMM K (for LDS B-panel)

typedef __attribute__((ext_vector_type(16))) _Float16 v16h;
typedef __attribute__((ext_vector_type(8)))  _Float16 v8h;
typedef __attribute__((ext_vector_type(8)))  float    v8f;
typedef __attribute__((ext_vector_type(4)))  unsigned int u32x4;
typedef __attribute__((ext_vector_type(8)))  unsigned int u32x8;

// ---------------------------------------------------------------------------
// Tensor Data Mover: DMA a contiguous region (ndwords DWORDs) from global
// memory into LDS at byte offset lds_off. D# built per CDNA5 ISA §8.3/8.4:
//   group0: count=1 | lds_addr | global_addr[56:0] | type=2 ("image")
//   group1: workgroup_mask=0 (not in cluster), data_size=2 (4B),
//           1-D tile: tensor_dim0 = tile_dim0 = ndwords, tensor_dim1=tile_dim1=1,
//           tensor_dim0_stride = ndwords. 2-D descriptor -> VADDR2/3 omitted.
// Completion tracked with TENSORcnt.
// ---------------------------------------------------------------------------
__device__ __forceinline__ void tdm_load_to_lds(const void* gsrc, unsigned lds_off,
                                                unsigned ndwords)
{
    const unsigned long long ga = (unsigned long long)(uintptr_t)gsrc;
    u32x4 g0;
    g0[0] = 1u;                                                 // count=1, user mode
    g0[1] = lds_off;                                            // lds_addr (bytes)
    g0[2] = (unsigned)(ga & 0xFFFFFFFFu);                       // global_addr[31:0]
    g0[3] = (unsigned)((ga >> 32) & 0x01FFFFFFu) | (2u << 30);  // addr[56:32] | type=2
    u32x8 g1;
    g1[0] = (2u << 16);                                  // wg_mask=0, data_size=4B
    g1[1] = (ndwords & 0xFFFFu) << 16;                   // tensor_dim0[15:0] @bits63:48
    g1[2] = ((ndwords >> 16) & 0xFFFFu) | (1u << 16);    // tensor_dim0[31:16], tensor_dim1=1
    g1[3] = (ndwords & 0xFFFFu) << 16;                   // tensor_dim1 hi=0, tile_dim0
    g1[4] = 1u;                                          // tile_dim1=1, tile_dim2=0
    g1[5] = ndwords;                                     // tensor_dim0_stride[31:0]
    g1[6] = 0u;                                          // stride0 hi, stride1 lo
    g1[7] = 0u;                                          // stride1 hi
    asm volatile("tensor_load_to_lds %0, %1" :: "s"(g0), "s"(g1) : "memory");
    __builtin_amdgcn_s_wait_tensorcnt(0);
}

// ---------------------------------------------------------------------------
// Patchify: pixels (B,3,512,512) -> A (2048, 768) f16, col index = c*256+py*16+px
// ---------------------------------------------------------------------------
__global__ __launch_bounds__(256)
void patchify_k(const float* __restrict__ px, _Float16* __restrict__ Ap)
{
    const int i = blockIdx.x * blockDim.x + threadIdx.x;
    if (i >= MM * 3 * PS * PS) return;
    const int col = i % (3 * PS * PS);
    const int row = i / (3 * PS * PS);
    const int b  = row / NN;
    const int n  = row % NN;
    const int gy = n / GG, gx = n % GG;
    const int c   = col / (PS * PS);
    const int rem = col % (PS * PS);
    const int py  = rem / PS, pxi = rem % PS;
    Ap[i] = (_Float16)px[(((size_t)(b * 3 + c)) * IMG + gy * PS + py) * IMG + gx * PS + pxi];
}

// ---------------------------------------------------------------------------
// f32 -> f16 cast
// ---------------------------------------------------------------------------
__global__ __launch_bounds__(256)
void cast_h_k(const float* __restrict__ in, _Float16* __restrict__ out, int n)
{
    const int i = blockIdx.x * blockDim.x + threadIdx.x;
    if (i < n) out[i] = (_Float16)in[i];
}

// ---------------------------------------------------------------------------
// WMMA GEMM: C(M,Nout) = A(M,K) @ W(Nout,K)^T + bias + optional residual
//   A, W f16 row-major; C f32. residual: res[(row % res_mod)*Nout + col].
// Block = 8 waves; all waves share one 16-wide n-tile, so the 16xK B-panel
// (contiguous rows of W) is staged into LDS ONCE per block by the Tensor Data
// Mover, then each wave reads its B fragments with ds_load_b128 while A
// fragments stream from global. One wave = one 16x16 tile via
// v_wmma_f32_16x16x32_f16 (f32 accumulate).
// Requires: M%128==0, Nout%16==0, K%64==0, K<=KMAX. All call sites satisfy.
// ---------------------------------------------------------------------------
__global__ __launch_bounds__(256)
void gemm_wmma_f16(const _Float16* __restrict__ A, const _Float16* __restrict__ W,
                   const float* __restrict__ bias, const float* __restrict__ res,
                   int res_mod, float* __restrict__ C, int M, int Nout, int K)
{
    __shared__ _Float16 sB[16 * KMAX];          // 24 KB B-panel

    const int lane   = threadIdx.x & 31;
    const int wave   = threadIdx.x >> 5;
    const int half16 = lane & 15;
    const int g      = lane >> 4;               // lane group: 0 (0-15), 1 (16-31)
    const int nt     = blockIdx.x;              // 16-wide tile column
    const int mt     = blockIdx.y * 8 + wave;   // 16-tall tile row

    // TDM: stage the contiguous 16-row B panel (16*K halves) into LDS (wave 0).
    if (threadIdx.x < 32) {
        tdm_load_to_lds(W + (size_t)nt * 16 * K,
                        (unsigned)(uintptr_t)(void*)sB,   // LDS byte offset
                        (unsigned)(16 * K * 2 / 4));      // DWORD count
    }
    __syncthreads();

    const _Float16* __restrict__ arow = A + (size_t)(mt * 16 + half16) * K; // A row M
    const _Float16* __restrict__ brow = sB + half16 * K;                    // B column N

    v8f acc = {};
    for (int k0 = 0; k0 < K; k0 += 64) {
        // A fragment (16x32 f16): lane holds row M=half16; half j -> K=k0+16g+j
        // (contiguous order, matching B)
        v8h al0 = *(const v8h*)(arow + k0 + g * 16);
        v8h ah0 = *(const v8h*)(arow + k0 + 8 + g * 16);
        v8h al1 = *(const v8h*)(arow + k0 + 32 + g * 16);
        v8h ah1 = *(const v8h*)(arow + k0 + 40 + g * 16);
        // B fragment (32x16 f16) from LDS: lane = column N=half16; half j -> K=k0+16g+j
        v16h b0 = *(const v16h*)(brow + k0 + g * 16);
        v16h b1 = *(const v16h*)(brow + k0 + 32 + g * 16);

        if (k0 + 64 < K) {   // CDNA5 global_prefetch_b8 for the next A K-slab
            __builtin_prefetch(arow + k0 + 64, 0, 1);
        }

        v16h a0, a1;
#pragma unroll
        for (int j = 0; j < 8; ++j) {
            a0[j] = al0[j]; a0[j + 8] = ah0[j];
            a1[j] = al1[j]; a1[j + 8] = ah1[j];
        }
        acc = __builtin_amdgcn_wmma_f32_16x16x32_f16(false, a0, false, b0, (short)0, acc, false, false);
        acc = __builtin_amdgcn_wmma_f32_16x16x32_f16(false, a1, false, b1, (short)0, acc, false, false);
    }

    // C/D layout: acc[r] <-> row = mt*16 + 8g + r, col = nt*16 + half16
    const int col = nt * 16 + half16;
    const float bv = bias ? bias[col] : 0.0f;
#pragma unroll
    for (int r = 0; r < 8; ++r) {
        const int rr = mt * 16 + g * 8 + r;
        float v = acc[r] + bv;
        if (res) v += res[(size_t)(rr % res_mod) * Nout + col];
        C[(size_t)rr * Nout + col] = v;
    }
}

// ---------------------------------------------------------------------------
// LayerNorm over D=384, one wave32 per row; writes f32 and/or f16.
// ---------------------------------------------------------------------------
__global__ __launch_bounds__(256)
void layernorm_k(const float* __restrict__ x, const float* __restrict__ w,
                 const float* __restrict__ b, float* __restrict__ outf,
                 _Float16* __restrict__ outh, int rows)
{
    const int lane = threadIdx.x & 31;
    const int row  = blockIdx.x * 8 + (threadIdx.x >> 5);
    if (row >= rows) return;
    const float* xr = x + (size_t)row * DD;
    float s = 0.f, s2 = 0.f;
    for (int j = lane; j < DD; j += 32) { float v = xr[j]; s += v; s2 += v * v; }
#pragma unroll
    for (int m = 16; m; m >>= 1) { s += __shfl_xor(s, m, 32); s2 += __shfl_xor(s2, m, 32); }
    const float mean = s * (1.0f / DD);
    const float var  = s2 * (1.0f / DD) - mean * mean;
    const float inv  = rsqrtf(var + 1e-5f);
    for (int j = lane; j < DD; j += 32) {
        const float v = (xr[j] - mean) * inv * w[j] + b[j];
        if (outf) outf[(size_t)row * DD + j] = v;
        if (outh) outh[(size_t)row * DD + j] = (_Float16)v;
    }
}

// ---------------------------------------------------------------------------
// Depthwise causal conv (K=4, left pad 3) over sequence + SiLU.
// x_main = xz[:, :, 0:E]. Writes f32 + f16.
// ---------------------------------------------------------------------------
__global__ __launch_bounds__(256)
void conv_silu_k(const float* __restrict__ xz, const float* __restrict__ cw,
                 const float* __restrict__ cb, float* __restrict__ xcf,
                 _Float16* __restrict__ xch)
{
    const int i = blockIdx.x * blockDim.x + threadIdx.x;
    if (i >= MM * EE) return;
    const int e  = i % EE;
    const int bn = i / EE;
    const int n  = bn % NN;
    float acc = cb[e];
#pragma unroll
    for (int k = 0; k < KK; ++k) {
        const int nn = n - (KK - 1) + k;
        if (nn >= 0)
            acc += cw[e * KK + k] * xz[((size_t)(bn - n + nn)) * (2 * EE) + e];
    }
    const float sv = acc * (1.f / (1.f + __expf(-acc)));   // silu
    xcf[i] = sv;
    xch[i] = (_Float16)sv;
}

// ---------------------------------------------------------------------------
// Fused selective scan: per (b,e) thread keeps h[16] in VGPRs, loops n=0..1023.
// Fuses: dt = clip(softplus(xB @ dt_w^T + dt_b)), dA = exp(clip(dt*A,-20,0)),
//        h = dA*h + xB*xc, y = <h,xC>, out = (y + Dp*xc)*silu(z) -> f16.
// xB/xC (32 floats per step) broadcast through LDS.
// ---------------------------------------------------------------------------
__global__ __launch_bounds__(256)
void scan_k(const float* __restrict__ xBC, const float* __restrict__ xz,
            const float* __restrict__ xc, const float* __restrict__ dt_w,
            const float* __restrict__ dt_b, const float* __restrict__ Aw,
            const float* __restrict__ Dp, _Float16* __restrict__ yh)
{
    __shared__ float sbc[2 * SS];
    const int e = blockIdx.x * blockDim.x + threadIdx.x;   // channel
    const int b = blockIdx.y;                              // batch
    float h[SS], dtw[SS], Aa[SS];
#pragma unroll
    for (int s = 0; s < SS; ++s) {
        h[s]   = 0.f;
        dtw[s] = dt_w[(size_t)e * SS + s];
        Aa[s]  = Aw[(size_t)e * SS + s];
    }
    const float dtb = dt_b[e];
    const float dpe = Dp[e];
    for (int n = 0; n < NN; ++n) {
        const size_t bn = (size_t)(b * NN + n);
        if (threadIdx.x < 2 * SS) sbc[threadIdx.x] = xBC[bn * (2 * SS) + threadIdx.x];
        __syncthreads();
        float dtv = dtb;
#pragma unroll
        for (int s = 0; s < SS; ++s) dtv += sbc[s] * dtw[s];
        dtv = (dtv > 20.f) ? dtv : log1pf(__expf(dtv));     // softplus
        dtv = fminf(fmaxf(dtv, 1e-4f), 1.0f);               // clip
        const float xcv = xc[bn * EE + e];
        float yv = 0.f;
#pragma unroll
        for (int s = 0; s < SS; ++s) {
            const float da = __expf(fminf(fmaxf(dtv * Aa[s], -20.f), 0.f));
            h[s] = da * h[s] + sbc[s] * xcv;
            yv  += h[s] * sbc[SS + s];
        }
        const float z = xz[bn * (2 * EE) + EE + e];
        const float outv = (yv + dpe * xcv) * (z * (1.f / (1.f + __expf(-z))));
        yh[bn * EE + e] = (_Float16)outv;
        __syncthreads();
    }
}

// ---------------------------------------------------------------------------
// Host orchestration
// ---------------------------------------------------------------------------
extern "C" void kernel_launch(void* const* d_in, const int* in_sizes, int n_in,
                              void* d_out, int out_size, void* d_ws, size_t ws_size,
                              hipStream_t stream)
{
    (void)in_sizes; (void)n_in; (void)out_size; (void)ws_size;

    const float* pixel   = (const float*)d_in[0];
    const float* patch_w = (const float*)d_in[1];
    const float* patch_b = (const float*)d_in[2];
    const float* pos     = (const float*)d_in[3];
    const float* norm_w  = (const float*)d_in[4];
    const float* norm_b  = (const float*)d_in[5];
    const float* in_w    = (const float*)d_in[6];
    const float* in_b    = (const float*)d_in[7];
    const float* conv_w  = (const float*)d_in[8];
    const float* conv_b  = (const float*)d_in[9];
    const float* xp_w    = (const float*)d_in[10];
    const float* xp_b    = (const float*)d_in[11];
    const float* dt_w    = (const float*)d_in[12];
    const float* dt_b    = (const float*)d_in[13];
    const float* Aw      = (const float*)d_in[14];
    const float* Dp      = (const float*)d_in[15];
    const float* out_w   = (const float*)d_in[16];
    const float* out_b   = (const float*)d_in[17];
    const float* fnorm_w = (const float*)d_in[18];
    const float* fnorm_b = (const float*)d_in[19];
    const float* scale_w = (const float*)d_in[20];
    const float* scale_b = (const float*)d_in[21];
    float* outp = (float*)d_out;

    // Workspace carve (256B aligned slabs)
    char* base = (char*)d_ws;
    size_t off = 0;
    auto carve = [&](size_t bytes) -> void* {
        void* p = base + off;
        off = (off + bytes + 255) & ~(size_t)255;
        return p;
    };
    _Float16* Ap   = (_Float16*)carve((size_t)MM * 768 * 2);          // patchified pixels f16
    _Float16* wP_h = (_Float16*)carve((size_t)DD * 768 * 2);          // patch_w f16
    _Float16* wIn  = (_Float16*)carve((size_t)LL * 2 * EE * DD * 2);  // in_w f16
    _Float16* wXp  = (_Float16*)carve((size_t)LL * 2 * SS * EE * 2);  // xp_w f16
    _Float16* wOut = (_Float16*)carve((size_t)LL * DD * EE * 2);      // out_w f16
    _Float16* wScl = (_Float16*)carve((size_t)3 * DD * DD * 2);       // scale_w f16
    _Float16* xh   = (_Float16*)carve((size_t)MM * DD * 2);           // f16 GEMM input
    _Float16* xch  = (_Float16*)carve((size_t)MM * EE * 2);           // f16 conv output
    _Float16* yh   = (_Float16*)carve((size_t)MM * EE * 2);           // f16 scan output
    float* x    = (float*)carve((size_t)MM * DD * 4);                 // residual stream f32
    float* xz   = (float*)carve((size_t)MM * 2 * EE * 4);             // in-proj output
    float* xcf  = (float*)carve((size_t)MM * EE * 4);                 // conv output f32
    float* xBCb = (float*)carve((size_t)MM * 2 * SS * 4);             // x-proj output

    auto cgrid = [](size_t n) { return dim3((unsigned)((n + 255) / 256)); };

    // One-time (per launch) weight casts to f16
    cast_h_k<<<cgrid((size_t)DD * 768), 256, 0, stream>>>(patch_w, wP_h, DD * 768);
    cast_h_k<<<cgrid((size_t)LL * 2 * EE * DD), 256, 0, stream>>>(in_w, wIn, LL * 2 * EE * DD);
    cast_h_k<<<cgrid((size_t)LL * 2 * SS * EE), 256, 0, stream>>>(xp_w, wXp, LL * 2 * SS * EE);
    cast_h_k<<<cgrid((size_t)LL * DD * EE), 256, 0, stream>>>(out_w, wOut, LL * DD * EE);
    cast_h_k<<<cgrid((size_t)3 * DD * DD), 256, 0, stream>>>(scale_w, wScl, 3 * DD * DD);
    patchify_k<<<cgrid((size_t)MM * 768), 256, 0, stream>>>(pixel, Ap);

    // Patch embed: x = Ap @ patch_w^T + patch_b + pos  (pos broadcast -> res_mod=N)
    gemm_wmma_f16<<<dim3(DD / 16, MM / 128), 256, 0, stream>>>(
        Ap, wP_h, patch_b, pos, NN, x, MM, DD, 3 * PS * PS);

    for (int i = 0; i < LL; ++i) {
        // LayerNorm -> f16 GEMM input
        layernorm_k<<<dim3(MM / 8), 256, 0, stream>>>(
            x, norm_w + i * DD, norm_b + i * DD, (float*)nullptr, xh, MM);
        // in-proj: xz = LN(x) @ in_w^T + in_b   (2048 x 1536 x 384)
        gemm_wmma_f16<<<dim3((2 * EE) / 16, MM / 128), 256, 0, stream>>>(
            xh, wIn + (size_t)i * 2 * EE * DD, in_b + i * 2 * EE, nullptr, 1, xz, MM, 2 * EE, DD);
        // depthwise causal conv + silu
        conv_silu_k<<<cgrid((size_t)MM * EE), 256, 0, stream>>>(
            xz, conv_w + i * EE * KK, conv_b + i * EE, xcf, xch);
        // x-proj: xBC = xc @ xp_w^T + xp_b      (2048 x 32 x 768)
        gemm_wmma_f16<<<dim3((2 * SS) / 16, MM / 128), 256, 0, stream>>>(
            xch, wXp + (size_t)i * 2 * SS * EE, xp_b + i * 2 * SS, nullptr, 1, xBCb, MM, 2 * SS, EE);
        // fused dt + selective scan + gating -> yh (f16)
        scan_k<<<dim3(EE / 256, BB), 256, 0, stream>>>(
            xBCb, xz, xcf, dt_w + (size_t)i * EE * SS, dt_b + i * EE,
            Aw + (size_t)i * EE * SS, Dp + i * EE, yh);
        // out-proj with residual: x = x + y @ out_w^T + out_b   (2048 x 384 x 768)
        gemm_wmma_f16<<<dim3(DD / 16, MM / 128), 256, 0, stream>>>(
            yh, wOut + (size_t)i * DD * EE, out_b + i * DD, x, MM, x, MM, DD, EE);
        // scale heads after layers 0, 4, 8
        if (i % 4 == 0 && i / 4 < 3) {
            const int j = i / 4;
            cast_h_k<<<cgrid((size_t)MM * DD), 256, 0, stream>>>(x, xh, MM * DD);
            gemm_wmma_f16<<<dim3(DD / 16, MM / 128), 256, 0, stream>>>(
                xh, wScl + (size_t)j * DD * DD, scale_b + j * DD, nullptr, 1,
                outp + (size_t)(1 + j) * MM * DD, MM, DD, DD);
        }
    }
    // Final LayerNorm -> output slot 0 (f32)
    layernorm_k<<<dim3(MM / 8), 256, 0, stream>>>(
        x, fnorm_w, fnorm_b, outp, (_Float16*)nullptr, MM);
}
